// Attention_76020921139691
// MI455X (gfx1250) — matrix operand order repaired and checked by reading, not compiled, not run on hardware
//
#include <hip/hip_runtime.h>
#include <hip/hip_bf16.h>
#include <math.h>

// ---------------------------------------------------------------------------
// Gated attention (Evoformer-style) for MI455X / gfx1250, wave32 + WMMA f16.
//   b=4, n=1024, D=1024, H=8, DH=64, INNER=512
// GEMMs use v_wmma_f32_16x16x32_f16; attention K/V tiles are staged into LDS
// with gfx1250 async DMA (global_load_async_to_lds_b128 / s_wait_asynccnt),
// double-buffered and shared by the 4 waves of each block.
// ---------------------------------------------------------------------------

typedef __attribute__((ext_vector_type(16))) _Float16 v16h;
typedef __attribute__((ext_vector_type(8)))  _Float16 v8h;
typedef __attribute__((ext_vector_type(8)))  float    v8f;

union Frag16 { v16h v; v8h h[2]; _Float16 e[16]; };
union Frag8f { v8f  v; float e[8]; };

static constexpr int kB     = 4;
static constexpr int kN     = 1024;
static constexpr int kD     = 1024;
static constexpr int kH     = 8;
static constexpr int kDH    = 64;
static constexpr int kInner = kH * kDH;       // 512
static constexpr int kM     = kB * kN;        // 4096 total rows

#define WMMA_F16(A, Bf, C) \
  __builtin_amdgcn_wmma_f32_16x16x32_f16(false, (A), false, (Bf), (short)0, (C), false, false)

// gfx1250 async DMA: LDS[vdst_lane] = MEM[vaddr_lane], tracked by ASYNCcnt.
__device__ __forceinline__ void async_ld_b128(unsigned int lds_off, const void* g) {
  asm volatile("global_load_async_to_lds_b128 %0, %1, off"
               :: "v"(lds_off), "v"((unsigned long long)(uintptr_t)g)
               : "memory");
}
template <int N>
__device__ __forceinline__ void wait_asynccnt() {
  asm volatile("s_wait_asynccnt %0" :: "i"(N) : "memory");
}
// Shared-memory pointer -> LDS byte offset (flat LDS aperture: addr[31:0]).
__device__ __forceinline__ unsigned int lds_off(const void* p) {
  return (unsigned int)(uintptr_t)p;
}

// ---------------------------------------------------------------------------
// Prep: f32 -> f16 convert, and transpose-convert (weights become [N][K]).
// ---------------------------------------------------------------------------
__global__ void k_cvt(const float* __restrict__ s, _Float16* __restrict__ d, int n) {
  int i = blockIdx.x * 256 + threadIdx.x;
  if (i < n) d[i] = (_Float16)s[i];
}

__global__ void k_transcvt(const float* __restrict__ s, _Float16* __restrict__ d,
                           int rows, int cols) {
  int i = blockIdx.x * 256 + threadIdx.x;
  if (i < rows * cols) {
    int r = i / cols, c = i % cols;
    d[c * rows + r] = (_Float16)s[i];  // [cols][rows] = W^T, K-contiguous
  }
}

// ---------------------------------------------------------------------------
// Fused projection GEMM: X[4096,1024] x {Wq|Wkv|Wg} -> Q,K (row-major),
// V (transposed [bh][dh][n]), gate logits (+bg, f32).
// One wave computes one 16x16 output tile. Ntot = 512(q)+1024(kv)+512(g).
// ---------------------------------------------------------------------------
__global__ __launch_bounds__(128) void k_proj(
    const _Float16* __restrict__ xh,
    const _Float16* __restrict__ wqt,  const _Float16* __restrict__ wkvt,
    const _Float16* __restrict__ wgt,  const float* __restrict__ bg,
    _Float16* __restrict__ qh, _Float16* __restrict__ kh,
    _Float16* __restrict__ vt, float* __restrict__ glog) {
  const int lane = threadIdx.x & 31;
  const int wid  = blockIdx.x * 4 + (threadIdx.x >> 5);
  const int NT   = 2048 / 16;                 // 128 column tiles
  const int tm = wid / NT, tn = wid % NT;
  const int hi = lane >> 4, ln = lane & 15;
  const int m0 = tm * 16;
  const int no = tn * 16 + ln;                // global output column

  const _Float16* bt; int brow;
  if (no < 512)       { bt = wqt;  brow = no; }
  else if (no < 1536) { bt = wkvt; brow = no - 512; }
  else                { bt = wgt;  brow = no - 1536; }
  const _Float16* bp = bt + brow * kD;
  const _Float16* ap = xh + (m0 + ln) * kD;

  Frag8f acc;
#pragma unroll
  for (int v = 0; v < 8; ++v) acc.e[v] = 0.f;

  for (int k0 = 0; k0 < kD; k0 += 32) {
    Frag16 a, b;
    a.h[0] = *(const v8h*)(ap + k0 + hi * 8);        // K 0..7  / 8..15
    a.h[1] = *(const v8h*)(ap + k0 + hi * 8 + 16);   // K 16..23/ 24..31
    b.h[0] = *(const v8h*)(bp + k0 + hi * 8);        // K 0..7  / 8..15
    b.h[1] = *(const v8h*)(bp + k0 + hi * 8 + 16);   // K 16..23/ 24..31
    acc.v = WMMA_F16(a.v, b.v, acc.v);
  }

  const float scale = 0.125f;  // DH^-0.5
#pragma unroll
  for (int v = 0; v < 8; ++v) {
    const int m  = m0 + hi * 8 + v;
    const int bb = m >> 10, i = m & 1023;
    const float val = acc.e[v];
    if (no < 512) {
      const int hh = no >> 6, dd = no & 63;
      qh[((bb * kH + hh) * kN + i) * kDH + dd] = (_Float16)(val * scale);
    } else if (no < 1024) {
      const int c = no - 512, hh = c >> 6, dd = c & 63;
      kh[((bb * kH + hh) * kN + i) * kDH + dd] = (_Float16)val;
    } else if (no < 1536) {
      const int c = no - 1024, hh = c >> 6, dd = c & 63;
      vt[((bb * kH + hh) * kDH + dd) * kN + i] = (_Float16)val;   // V^T
    } else {
      glog[m * kInner + (no - 1536)] = val + bg[no - 1536];
    }
  }
}

// ---------------------------------------------------------------------------
// Flash attention. Block = 4 waves = 4 consecutive 16-row query tiles of the
// SAME (b,h). Per 32-key chunk the block stages K (32x64 f16, contiguous) and
// V^T (64x32 f16, 64 strided 64B rows) into LDS via async DMA, double
// buffered; all 4 waves build WMMA fragments from LDS (ds_load_b128).
// Online softmax uses 16-lane shfl_xor reductions; P transposes C->A layout
// through a per-wave LDS scratch tile. Epilogue fuses 1/l + sigmoid gating.
// ---------------------------------------------------------------------------
__global__ __launch_bounds__(128) void k_attn(
    const _Float16* __restrict__ qh, const _Float16* __restrict__ kh,
    const _Float16* __restrict__ vt, const float* __restrict__ bias,
    const unsigned char* __restrict__ mask, const float* __restrict__ glog,
    _Float16* __restrict__ G) {
  __shared__ _Float16 ldsk[2][32 * 64];   // K chunk  [j 0..31][d 0..63]
  __shared__ _Float16 ldsv[2][64 * 32];   // V^T chunk [d 0..63][j 0..31]
  __shared__ _Float16 lds_p[4][16 * 32];  // per-wave P staging
  const int tid  = threadIdx.x;
  const int lane = tid & 31;
  const int w    = tid >> 5;
  const int wid  = blockIdx.x * 4 + w;
  const int it = wid & 63, bh = wid >> 6;   // bh in [0,32)
  const int bb = bh >> 3, hh = bh & 7;
  const int hi = lane >> 4, ln = lane & 15;
  const int i0 = it * 16;
  const _Float16* qbase = qh + (size_t)bh * kN * kDH;
  const _Float16* kbase = kh + (size_t)bh * kN * kDH;
  const _Float16* vbase = vt + (size_t)bh * kDH * kN;
  const float*    bbase = bias + (size_t)bh * kN * kN;
  _Float16* ldsp = lds_p[w];

  // --- async stage of one 32-key chunk (4 async b128 per thread => 4/wave) ---
  auto stage = [&](int jt, int buf) {
    // K: rows jt*32..+31 are contiguous 4KB; thread copies 32B.
    {
      const _Float16* g = kbase + jt * (32 * kDH) + tid * 16;
      const unsigned int l = lds_off(&ldsk[buf][0]) + tid * 32;
      async_ld_b128(l,      g);
      async_ld_b128(l + 16, g + 8);
    }
    // V^T: 64 rows of 64B (stride 2KB); thread (d = tid>>1, half = tid&1).
    {
      const int d = tid >> 1, half = tid & 1;
      const _Float16* g = vbase + d * kN + jt * 32 + half * 16;
      const unsigned int l = lds_off(&ldsv[buf][0]) + (d * 32 + half * 16) * 2;
      async_ld_b128(l,      g);
      async_ld_b128(l + 16, g + 8);
    }
  };

  Frag16 aq[2];
#pragma unroll
  for (int kk = 0; kk < 2; ++kk) {
    const _Float16* p = qbase + (i0 + ln) * kDH + kk * 32 + hi * 8;
    aq[kk].h[0] = *(const v8h*)(p);
    aq[kk].h[1] = *(const v8h*)(p + 16);
  }

  bool rmask[8];
#pragma unroll
  for (int v = 0; v < 8; ++v) rmask[v] = mask[bb * kN + i0 + hi * 8 + v] != 0;

  float mrow[8], lrow[8];
#pragma unroll
  for (int v = 0; v < 8; ++v) { mrow[v] = -INFINITY; lrow[v] = 0.f; }
  Frag8f o[4];
#pragma unroll
  for (int dc = 0; dc < 4; ++dc)
#pragma unroll
    for (int v = 0; v < 8; ++v) o[dc].e[v] = 0.f;

  stage(0, 0);
  for (int jt = 0; jt < kN / 32; ++jt) {
    const int buf = jt & 1;
    if (jt + 1 < kN / 32) {
      stage(jt + 1, buf ^ 1);   // overlap next tile's DMA with this compute
      wait_asynccnt<4>();       // in-order: first 4 (tile jt) are done
    } else {
      wait_asynccnt<0>();
    }
    __syncthreads();            // tile jt visible to all 4 waves

    float sv[2][8];
#pragma unroll
    for (int s = 0; s < 2; ++s) {
      const int j0 = jt * 32 + s * 16;
      Frag8f sf;
#pragma unroll
      for (int v = 0; v < 8; ++v) sf.e[v] = 0.f;
#pragma unroll
      for (int kk = 0; kk < 2; ++kk) {
        Frag16 bk;  // B fragment from LDS K chunk (row = local key)
        const _Float16* p = &ldsk[buf][(s * 16 + ln) * kDH + kk * 32 + hi * 8];
        bk.h[0] = *(const v8h*)(p);
        bk.h[1] = *(const v8h*)(p + 16);
        sf.v = WMMA_F16(aq[kk].v, bk.v, sf.v);
      }
      const bool cm = mask[bb * kN + j0 + ln] != 0;
      const float* bp = bbase + (i0 + hi * 8) * kN + j0 + ln;
#pragma unroll
      for (int v = 0; v < 8; ++v) {
        float val = sf.e[v] + bp[v * kN];
        if (!(cm && rmask[v])) val = -3.0e38f;
        sv[s][v] = val;
      }
    }
    if (jt + 1 < kN / 32)  // pull next bias tile toward L2/L0
      __builtin_prefetch(bbase + (i0 + hi * 8) * kN + (jt + 1) * 32 + ln, 0, 0);

    // Online softmax across the 32 new columns (row i lives on 16 lanes).
    float alpha[8];
#pragma unroll
    for (int v = 0; v < 8; ++v) {
      float t = fmaxf(sv[0][v], sv[1][v]);
#pragma unroll
      for (int x = 1; x < 16; x <<= 1) t = fmaxf(t, __shfl_xor(t, x, 32));
      const float mnew = fmaxf(mrow[v], t);
      alpha[v] = __expf(mrow[v] - mnew);
      mrow[v] = mnew;
      const float p0 = __expf(sv[0][v] - mnew);
      const float p1 = __expf(sv[1][v] - mnew);
      sv[0][v] = p0; sv[1][v] = p1;
      float rs = p0 + p1;
#pragma unroll
      for (int x = 1; x < 16; x <<= 1) rs += __shfl_xor(rs, x, 32);
      lrow[v] = lrow[v] * alpha[v] + rs;
    }
#pragma unroll
    for (int dc = 0; dc < 4; ++dc)
#pragma unroll
      for (int v = 0; v < 8; ++v) o[dc].e[v] *= alpha[v];

    // Stage P (C layout) into per-wave LDS as f16, reload in A layout.
#pragma unroll
    for (int v = 0; v < 8; ++v) {
      ldsp[(hi * 8 + v) * 32 + ln]      = (_Float16)sv[0][v];
      ldsp[(hi * 8 + v) * 32 + 16 + ln] = (_Float16)sv[1][v];
    }
    Frag16 ap;
    ap.h[0] = *(const v8h*)(ldsp + ln * 32 + hi * 8);
    ap.h[1] = *(const v8h*)(ldsp + ln * 32 + hi * 8 + 16);

#pragma unroll
    for (int dc = 0; dc < 4; ++dc) {
      Frag16 bv;  // B fragment from LDS V^T chunk (row = d)
      const _Float16* p = &ldsv[buf][(dc * 16 + ln) * 32 + hi * 8];
      bv.h[0] = *(const v8h*)(p);
      bv.h[1] = *(const v8h*)(p + 16);
      o[dc].v = WMMA_F16(ap.v, bv.v, o[dc].v);
    }
    __syncthreads();  // everyone done reading buf before it is re-staged
  }

  // Epilogue: 1/l, sigmoid gate, f16 store to G.
#pragma unroll
  for (int v = 0; v < 8; ++v) {
    const float invl = 1.f / lrow[v];
    const int gm = bb * kN + i0 + hi * 8 + v;
#pragma unroll
    for (int dc = 0; dc < 4; ++dc) {
      const int gc = hh * kDH + dc * 16 + ln;
      const float g = 1.f / (1.f + __expf(-glog[gm * kInner + gc]));
      G[gm * kInner + gc] = (_Float16)(o[dc].e[v] * invl * g);
    }
  }
}

// ---------------------------------------------------------------------------
// Output GEMM: Y[4096,1024] = G[4096,512] @ Wo + bo  (f32 out)
// ---------------------------------------------------------------------------
__global__ __launch_bounds__(128) void k_out(
    const _Float16* __restrict__ G, const _Float16* __restrict__ wot,
    const float* __restrict__ bo, float* __restrict__ y) {
  const int lane = threadIdx.x & 31;
  const int wid  = blockIdx.x * 4 + (threadIdx.x >> 5);
  const int NT = kD / 16;  // 64
  const int tm = wid / NT, tn = wid % NT;
  const int hi = lane >> 4, ln = lane & 15;
  const _Float16* ap = G   + (tm * 16 + ln) * kInner;
  const _Float16* bp = wot + (tn * 16 + ln) * kInner;

  Frag8f acc;
#pragma unroll
  for (int v = 0; v < 8; ++v) acc.e[v] = 0.f;
  for (int k0 = 0; k0 < kInner; k0 += 32) {
    Frag16 a, b;
    a.h[0] = *(const v8h*)(ap + k0 + hi * 8);
    a.h[1] = *(const v8h*)(ap + k0 + hi * 8 + 16);
    b.h[0] = *(const v8h*)(bp + k0 + hi * 8);
    b.h[1] = *(const v8h*)(bp + k0 + hi * 8 + 16);
    acc.v = WMMA_F16(a.v, b.v, acc.v);
  }
  const float bb = bo[tn * 16 + ln];
#pragma unroll
  for (int v = 0; v < 8; ++v) {
    const int m = tm * 16 + hi * 8 + v;
    y[m * kD + tn * 16 + ln] = acc.e[v] + bb;
  }
}

// ---------------------------------------------------------------------------
// Host launcher. Inputs: x, mask, attn_bias, Wq, Wkv, Wg, bg, Wo, bo.
// ---------------------------------------------------------------------------
extern "C" void kernel_launch(void* const* d_in, const int* in_sizes, int n_in,
                              void* d_out, int out_size, void* d_ws, size_t ws_size,
                              hipStream_t stream) {
  const float* x    = (const float*)d_in[0];
  const unsigned char* mask = (const unsigned char*)d_in[1];  // jnp bool -> 1 byte
  const float* bias = (const float*)d_in[2];
  const float* Wq   = (const float*)d_in[3];
  const float* Wkv  = (const float*)d_in[4];
  const float* Wg   = (const float*)d_in[5];
  const float* bg   = (const float*)d_in[6];
  const float* Wo   = (const float*)d_in[7];
  const float* bo   = (const float*)d_in[8];
  float* y = (float*)d_out;

  // Workspace carve-up (all f16 chunks have even element counts -> 4B align).
  _Float16* xh   = (_Float16*)d_ws;                 // 4096*1024
  _Float16* wqt  = xh   + (size_t)kM * kD;          //  512*1024
  _Float16* wkvt = wqt  + (size_t)512 * kD;         // 1024*1024
  _Float16* wgt  = wkvt + (size_t)1024 * kD;        //  512*1024
  _Float16* wot  = wgt  + (size_t)512 * kD;         // 1024*512
  _Float16* qh   = wot  + (size_t)kD * kInner;      // 4*8*1024*64
  _Float16* kh   = qh   + (size_t)kB * kH * kN * kDH;
  _Float16* vt   = kh   + (size_t)kB * kH * kN * kDH;
  _Float16* G    = vt   + (size_t)kB * kH * kN * kDH;  // 4096*512
  float*    glog = (float*)(G + (size_t)kM * kInner);  // 4096*512 f32

  // 1) precision conversions / weight transposes
  k_cvt<<<(kM * kD + 255) / 256, 256, 0, stream>>>(x, xh, kM * kD);
  k_transcvt<<<(kD * kInner     + 255) / 256, 256, 0, stream>>>(Wq,  wqt,  kD, kInner);
  k_transcvt<<<(kD * 2 * kInner + 255) / 256, 256, 0, stream>>>(Wkv, wkvt, kD, 2 * kInner);
  k_transcvt<<<(kD * kInner     + 255) / 256, 256, 0, stream>>>(Wg,  wgt,  kD, kInner);
  k_transcvt<<<(kInner * kD     + 255) / 256, 256, 0, stream>>>(Wo,  wot,  kInner, kD);

  // 2) fused QKV + gate-logit projection: (4096/16)*(2048/16)=32768 waves
  k_proj<<<32768 / 4, 128, 0, stream>>>(xh, wqt, wkvt, wgt, bg, qh, kh, vt, glog);

  // 3) flash attention + gating: 4*8*64 = 2048 waves (512 blocks x 4 waves)
  k_attn<<<2048 / 4, 128, 0, stream>>>(qh, kh, vt, bias, mask, glog, G);

  // 4) output GEMM: (4096/16)*(1024/16)=16384 waves
  k_out<<<16384 / 4, 128, 0, stream>>>(G, wot, bo, y);
}
